// tm_3513283248662
// MI455X (gfx1250) — compile-verified, matrix operand order checked by reading, not run on hardware
//
#include <hip/hip_runtime.h>
#include <hip/hip_bf16.h>

typedef __attribute__((ext_vector_type(16))) _Float16 v16h;
typedef __attribute__((ext_vector_type(8)))  float    v8f;

#define SEQ 118
#define SP  128   // padded sequence / row stride
#define DD  128   // model dim
#define NH  4
#define DH  32
#define NT  256   // threads per block (8 waves)

union FragH { v16h v; _Float16 h[16]; unsigned int w[8]; };
union FragF { v8f  v; float f[8]; };

// K offset inside a 16x32 f16 fragment for VGPR pair p (0..7), half-wave hs.
__device__ __forceinline__ int kb_of(int p, int hs) {
  return (p < 4) ? (hs * 8 + 2 * p) : (16 + hs * 8 + 2 * (p - 4));
}

// 16x32 f16 fragment from LDS in [row][K] layout (row stride SP, K contiguous).
// Used for A tiles (row = M) and for B tiles staged transposed as [N][K] (row = N).
__device__ __forceinline__ v16h load_rk(const _Float16* base, int r0, int k0, int lane) {
  int row = r0 + (lane & 15);
  int hs  = (lane >> 4) & 1;
  const _Float16* rp = base + row * SP + k0;
  FragH u;
#pragma unroll
  for (int p = 0; p < 8; ++p)
    u.w[p] = *(const unsigned int*)(rp + kb_of(p, hs));   // 2 halves / 32-bit ds load
  return u.v;
}

// Generic B fragment: element (K,n) at base[(k0+K)*rs + n*cs].
__device__ __forceinline__ v16h load_b_gen(const _Float16* base, int k0, int n0,
                                           int rs, int cs, int lane) {
  int col = n0 + (lane & 15);
  int hs  = (lane >> 4) & 1;
  FragH u;
#pragma unroll
  for (int p = 0; p < 8; ++p) {
    int kb = kb_of(p, hs);
    u.h[2 * p]     = base[(k0 + kb)     * rs + col * cs];
    u.h[2 * p + 1] = base[(k0 + kb + 1) * rs + col * cs];
  }
  return u.v;
}

__device__ __forceinline__ v8f wmma_f16(v16h a, v16h b, v8f c) {
  return __builtin_amdgcn_wmma_f32_16x16x32_f16(false, a, false, b, (short)0, c, false, false);
}

// One attention block: hA (f16 activations) -> x1 (f32), optional residual add.
// Scratch reuse: hA doubles as the scores/probs buffer (dead after qkv GEMM);
// Y is written into consumed q columns of qF; fc2 weights staged into kF.
__device__ __forceinline__ void attn_block(
    _Float16* hA, _Float16* qF, _Float16* kF, _Float16* vF, float* x1,
    const _Float16* qkvWh, float* red,
    const float* cQkvB, const float* cFc2B,
    const float* __restrict__ fc2_w,
    int tid, int lane, int wave, bool add_residual)
{
  // ---------- qkv = hA @ qkv_w^T + qkv_b  (M=128, N=384, K=128), all-LDS ----------
  for (int t = wave; t < 8 * 24; t += 8) {
    int m0 = (t / 24) * 16;           // wave-uniform (t derived from readfirstlane)
    int n0 = (t % 24) * 16;
    v8f c = {};
#pragma unroll
    for (int kk = 0; kk < DD; kk += 32)
      c = wmma_f16(load_rk(hA, m0, kk, lane), load_rk(qkvWh, n0, kk, lane), c);
    FragF acc; acc.v = c;
    // destination is uniform per 16-wide tile (n0 never straddles 128 boundaries)
    _Float16* dst; int nb;
    if (n0 < 128)      { dst = qF; nb = n0; }
    else if (n0 < 256) { dst = kF; nb = n0 - 128; }
    else               { dst = vF; nb = n0 - 256; }
    int nc   = nb + (lane & 15);
    int moff = ((lane >> 4) & 1) * 8;
    float bias = cQkvB[n0 + (lane & 15)];
#pragma unroll
    for (int r = 0; r < 8; ++r)
      dst[(m0 + moff + r) * SP + nc] = (_Float16)(acc.f[r] + bias);
  }
  __syncthreads();

  // ---------- RoPE on q, k (bf16-rounded tables, fp32 math) ----------
  for (int idx = tid; idx < SEQ * 64; idx += NT) {
    int s = idx / 64, ch = idx % 64;
    float fr = (float)s * __powf(10000.0f, -(float)ch * (1.0f / 64.0f));
    float cs = __bfloat162float(__float2bfloat16(__cosf(fr)));
    float sn = __bfloat162float(__float2bfloat16(__sinf(fr)));
    {
      float a1 = (float)qF[s * SP + ch], a2 = (float)qF[s * SP + ch + 64];
      qF[s * SP + ch]      = (_Float16)( a1 * cs + a2 * sn);
      qF[s * SP + ch + 64] = (_Float16)(-a1 * sn + a2 * cs);
    }
    {
      float a1 = (float)kF[s * SP + ch], a2 = (float)kF[s * SP + ch + 64];
      kF[s * SP + ch]      = (_Float16)( a1 * cs + a2 * sn);
      kF[s * SP + ch + 64] = (_Float16)(-a1 * sn + a2 * cs);
    }
  }
  __syncthreads();

  const float scale = 0.17677669529663687f;  // 1/sqrt(DH)
  for (int h = 0; h < NH; ++h) {
    int h0 = h * DH;
    // ---- scores = q_h @ k_h^T * scale  (M=128,N=128,K=32) -> hA as f16 ----
    for (int t = wave; t < 64; t += 8) {
      int m0 = (t >> 3) * 16, n0 = (t & 7) * 16;
      v8f c = {};
      c = wmma_f16(load_rk(qF, m0, h0, lane),
                   load_b_gen(kF + h0, 0, n0, /*rs=*/1, /*cs=*/SP, lane), c);
      FragF acc; acc.v = c;
      int n    = n0 + (lane & 15);
      int moff = ((lane >> 4) & 1) * 8;
#pragma unroll
      for (int r = 0; r < 8; ++r)
        hA[(m0 + moff + r) * SP + n] = (_Float16)(acc.f[r] * scale);
    }
    __syncthreads();

    // ---- causal softmax in place on hA: 2 threads per row, fp32 math ----
    {
      int r   = tid & 127;
      int hh  = tid >> 7;                    // which half of the row
      int lim = (r < SEQ) ? r : (SEQ - 1);   // valid keys: 0..lim
      int c0  = hh * 64, c1 = c0 + 64;
      int ce  = (c1 < lim + 1) ? c1 : (lim + 1);  // exclusive valid end

      float mx = -3.0e38f;
      for (int cc = c0; cc < ce; ++cc) mx = fmaxf(mx, (float)hA[r * SP + cc]);
      red[tid] = mx;
      __syncthreads();
      mx = fmaxf(red[r], red[r + 128]);
      __syncthreads();                       // reads of max done before sum writes

      float sum = 0.0f;
      for (int cc = c0; cc < ce; ++cc) {
        float e = __expf((float)hA[r * SP + cc] - mx);
        hA[r * SP + cc] = (_Float16)e;       // exp in place (thread owns segment)
        sum += e;
      }
      for (int cc = ce; cc < c1; ++cc) hA[r * SP + cc] = (_Float16)0.0f;
      red[tid] = sum;
      __syncthreads();
      float inv = 1.0f / (red[r] + red[r + 128]);
      for (int cc = c0; cc < ce; ++cc)
        hA[r * SP + cc] = (_Float16)((float)hA[r * SP + cc] * inv);
    }
    __syncthreads();

    // ---- y_h = P @ v_h (M=128,N=32,K=128) -> consumed q columns of qF ----
    for (int t = wave; t < 16; t += 8) {
      int m0 = (t >> 1) * 16, n0 = (t & 1) * 16;
      v8f c = {};
#pragma unroll
      for (int kk = 0; kk < SP; kk += 32)
        c = wmma_f16(load_rk(hA, m0, kk, lane),
                     load_b_gen(vF + h0, kk, n0, /*rs=*/SP, /*cs=*/1, lane), c);
      FragF acc; acc.v = c;
      int n    = h0 + n0 + (lane & 15);
      int moff = ((lane >> 4) & 1) * 8;
#pragma unroll
      for (int r = 0; r < 8; ++r) {
        int m = m0 + moff + r;
        qF[m * SP + n] = (m < SEQ) ? (_Float16)acc.f[r] : (_Float16)0.0f;
      }
    }
    __syncthreads();
  }

  // ---- stage fc2_w (f16, [N][K]) into kF's region (k is dead now) ----
  for (int i = tid * 4; i < DD * DD; i += NT * 4) {
    float4 w4 = *(const float4*)(fc2_w + i);
    union { _Float16 h[4]; unsigned int u[2]; } t;
    t.h[0] = (_Float16)w4.x; t.h[1] = (_Float16)w4.y;
    t.h[2] = (_Float16)w4.z; t.h[3] = (_Float16)w4.w;
    *(unsigned int*)(kF + i)     = t.u[0];
    *(unsigned int*)(kF + i + 2) = t.u[1];
  }
  __syncthreads();

  // ---------- x1 = [x1 +] silu(Y @ fc2_w^T + fc2_b)  (all-LDS) ----------
  for (int t = wave; t < 64; t += 8) {
    int m0 = (t >> 3) * 16, n0 = (t & 7) * 16;
    v8f c = {};
#pragma unroll
    for (int kk = 0; kk < DD; kk += 32)
      c = wmma_f16(load_rk(qF, m0, kk, lane), load_rk(kF, n0, kk, lane), c);
    FragF acc; acc.v = c;
    int n    = n0 + (lane & 15);
    int moff = ((lane >> 4) & 1) * 8;
    float bias = cFc2B[n];
#pragma unroll
    for (int r = 0; r < 8; ++r) {
      int m = m0 + moff + r;
      float z   = acc.f[r] + bias;
      float sil = z / (1.0f + __expf(-z));
      float o   = (m < SEQ) ? sil : 0.0f;
      x1[m * SP + n] = (add_residual ? x1[m * SP + n] : 0.0f) + o;
    }
  }
  __syncthreads();
}

extern "C" __global__ __launch_bounds__(NT)
void mini_transformer_kernel(const float* __restrict__ x,
                             const float* __restrict__ fc1_w,
                             const float* __restrict__ fc1_b,
                             const float* __restrict__ qkv_w,
                             const float* __restrict__ qkv_b,
                             const float* __restrict__ fc2_w,
                             const float* __restrict__ fc2_b,
                             const float* __restrict__ fc3_w,
                             const float* __restrict__ fc3_b,
                             const float* __restrict__ ln_g,
                             const float* __restrict__ ln_b,
                             float* __restrict__ out)
{
  extern __shared__ __align__(16) char smem[];
  float*    x1    = (float*)smem;                    // 64 KB  fp32 residual/output
  _Float16* qkvWh = (_Float16*)(smem + 65536);       // 96 KB  f16 qkv weights [N][K]
  _Float16* hA    = (_Float16*)(smem + 163840);      // 32 KB  f16 act / scores / probs
  _Float16* qF    = (_Float16*)(smem + 196608);      // 32 KB  f16 q, then Y
  _Float16* kF    = (_Float16*)(smem + 229376);      // 32 KB  f16 k, then fc2 weights
  _Float16* vF    = (_Float16*)(smem + 262144);      // 32 KB  f16 v
  float*    red   = (float*)(smem + 294912);         // 1 KB   softmax reductions
  float*    cQkvB = (float*)(smem + 295936);         // 1.5 KB qkv bias
  float*    cFc2B = (float*)(smem + 297472);         // 0.5 KB fc2 bias  (total ~291 KB)

  int tid  = threadIdx.x;
  int lane = tid & 31;
  // Physically uniform within a wave32; readfirstlane pins it to an SGPR so all
  // tile coordinates / LDS base selects become scalar (no exec-mask branching).
  int wave = __builtin_amdgcn_readfirstlane(tid >> 5);
  int b = blockIdx.x;
  const float* xrow = x + (size_t)b * SEQ;

  // ---- stage qkv_w to f16 LDS (persists across both blocks) + biases ----
  for (int i = tid * 4; i < 3 * DD * DD; i += NT * 4) {
    float4 w4 = *(const float4*)(qkv_w + i);
    union { _Float16 h[4]; unsigned int u[2]; } t;
    t.h[0] = (_Float16)w4.x; t.h[1] = (_Float16)w4.y;
    t.h[2] = (_Float16)w4.z; t.h[3] = (_Float16)w4.w;
    *(unsigned int*)(qkvWh + i)     = t.u[0];
    *(unsigned int*)(qkvWh + i + 2) = t.u[1];
  }
  for (int i = tid; i < 3 * DD; i += NT) cQkvB[i] = qkv_b[i];
  if (tid < DD) cFc2B[tid] = fc2_b[tid];

  // ---- h = silu(x[:,None] * fc1_w + fc1_b) -> hA (f16), zero pad rows ----
  for (int idx = tid; idx < SP * DD; idx += NT) {
    int s = idx >> 7, kk = idx & 127;
    float v = 0.0f;
    if (s < SEQ) {
      float z = xrow[s] * fc1_w[kk] + fc1_b[kk];
      v = z / (1.0f + __expf(-z));
    }
    hA[idx] = (_Float16)v;
  }
  __syncthreads();

  attn_block(hA, qF, kF, vF, x1, qkvWh, red, cQkvB, cFc2B, fc2_w,
             tid, lane, wave, /*add_residual=*/false);

  // ---- hA = layernorm(x1) (f16), zero pad rows ----
  if (tid < SP) {
    int s = tid;
    if (s < SEQ) {
      float mu = 0.0f;
      for (int k = 0; k < DD; ++k) mu += x1[s * SP + k];
      mu *= (1.0f / DD);
      float var = 0.0f;
      for (int k = 0; k < DD; ++k) { float d = x1[s * SP + k] - mu; var += d * d; }
      var *= (1.0f / DD);
      float rstd = rsqrtf(var + 1e-5f);
      for (int k = 0; k < DD; ++k)
        hA[s * SP + k] = (_Float16)((x1[s * SP + k] - mu) * rstd * ln_g[k] + ln_b[k]);
    } else {
      for (int k = 0; k < DD; ++k) hA[s * SP + k] = (_Float16)0.0f;
    }
  }
  __syncthreads();

  attn_block(hA, qF, kF, vF, x1, qkvWh, red, cQkvB, cFc2B, fc2_w,
             tid, lane, wave, /*add_residual=*/true);

  // ---- out = x2 @ fc3_w^T + fc3_b ----
  float b3 = fc3_b[0];
  for (int s = tid; s < SEQ; s += NT) {
    float acc = b3;
    for (int k = 0; k < DD; ++k) acc += x1[s * SP + k] * fc3_w[k];
    out[(size_t)b * SEQ + s] = acc;
  }
}

extern "C" void kernel_launch(void* const* d_in, const int* in_sizes, int n_in,
                              void* d_out, int out_size, void* d_ws, size_t ws_size,
                              hipStream_t stream)
{
  (void)n_in; (void)out_size; (void)d_ws; (void)ws_size;
  const float* x     = (const float*)d_in[0];
  const float* fc1_w = (const float*)d_in[1];
  const float* fc1_b = (const float*)d_in[2];
  const float* qkv_w = (const float*)d_in[3];
  const float* qkv_b = (const float*)d_in[4];
  const float* fc2_w = (const float*)d_in[5];
  const float* fc2_b = (const float*)d_in[6];
  const float* fc3_w = (const float*)d_in[7];
  const float* fc3_b = (const float*)d_in[8];
  const float* ln_g  = (const float*)d_in[9];
  const float* ln_b  = (const float*)d_in[10];
  float* out = (float*)d_out;

  int batch = in_sizes[0] / SEQ;              // 8192
  size_t lds_bytes = 297984;                  // ~291 KB dynamic LDS (<= 320 KB/WGP)
  hipFuncSetAttribute((const void*)mini_transformer_kernel,
                      hipFuncAttributeMaxDynamicSharedMemorySize, (int)lds_bytes);

  mini_transformer_kernel<<<batch, NT, lds_bytes, stream>>>(
      x, fc1_w, fc1_b, qkv_w, qkv_b, fc2_w, fc2_b, fc3_w, fc3_b, ln_g, ln_b, out);
}
